// MESPool_36988258353719
// MI455X (gfx1250) — matrix operand order repaired and checked by reading, not compiled, not run on hardware
//
#include <hip/hip_runtime.h>
#include <math.h>
#include <limits.h>

// ---------------------------------------------------------------------------
// MESPool pipeline for MI455X (gfx1250, wave32, WMMA).
// Dense D=128 GEMMs -> v_wmma_f32_16x16x32_f16 (f32 accumulate, f16 operands).
// GEMM activations (ea, h, hid) stored in f16 (consumed only by WMMA; f32
// would double HBM traffic on a bandwidth-bound pipeline).
// Plain-A GEMM strips staged into LDS with GLOBAL_LOAD_ASYNC_TO_LDS_B128
// (ASYNCcnt path); the unit-mix (K) GEMM stages through VGPRs with packed
// f16 math. GEMM modes are template parameters so every call site gets a
// branch-free, fully unrolled epilogue.
// ---------------------------------------------------------------------------

#define D_DIM 128
#define B_GR  64
#define N_NODE 40000
#define E_EDGE 400000
#define K_CHUNK 50000
#define INV_KEY 1600000000u
#define LN_EPS 1e-5f

typedef __attribute__((ext_vector_type(16))) _Float16 v16h;
typedef __attribute__((ext_vector_type(8)))  _Float16 v8h;
typedef __attribute__((ext_vector_type(4)))  _Float16 v4h;
typedef __attribute__((ext_vector_type(8)))  float    v8f;

union V16 { v16h v; v8h h[2]; };

static inline int cdiv(int a, int b) { return (a + b - 1) / b; }

// ------------------------------- utility ----------------------------------

__global__ void mes_fill_f(float* p, float v, int n) {
    int i = blockIdx.x * blockDim.x + threadIdx.x;
    if (i < n) p[i] = v;
}
__global__ void mes_fill_i(int* p, int v, int n) {
    int i = blockIdx.x * blockDim.x + threadIdx.x;
    if (i < n) p[i] = v;
}
__global__ void mes_cvt_f16(const float* __restrict__ src, _Float16* __restrict__ dst, int n) {
    int i = blockIdx.x * blockDim.x + threadIdx.x;
    if (i < n) dst[i] = (_Float16)src[i];
}

__device__ inline void atomicMaxFloat(float* addr, float v) {
    int* ai = (int*)addr;
    int old = __float_as_int(*addr);
    while (__int_as_float(old) < v) {
        int assumed = old;
        old = atomicCAS(ai, assumed, __float_as_int(v));
        if (old == assumed) break;
    }
}

// --------------------------- stage 1: msg / ea ------------------------------

__global__ void mes_edge_seg_sum(const float* __restrict__ eattr, const int* __restrict__ col,
                                 float* __restrict__ msum, float* __restrict__ cnt, int E) {
    int idx = blockIdx.x * blockDim.x + threadIdx.x;
    int e = idx >> 7, d = idx & 127;
    if (e >= E) return;
    atomicAdd(&msum[(size_t)col[e] * D_DIM + d], eattr[(size_t)e * D_DIM + d]);
    if (d == 0) atomicAdd(&cnt[col[e]], 1.0f);
}

__global__ void mes_msg_div(float* __restrict__ msum, const float* __restrict__ cnt, int N) {
    int idx = blockIdx.x * blockDim.x + threadIdx.x;
    int i = idx >> 7;
    if (i >= N) return;
    msum[idx] /= fmaxf(cnt[i], 1.0f);
}

// ea kept in f16: consumed only by the WMMA GEMMs.
__global__ void mes_ea(const float* __restrict__ eattr, const float* __restrict__ msum,
                       const float* __restrict__ x, const int* __restrict__ row,
                       _Float16* __restrict__ eaH, int E) {
    int idx = blockIdx.x * blockDim.x + threadIdx.x;
    int e = idx >> 7, d = idx & 127;
    if (e >= E) return;
    int r = row[e];
    float v = (eattr[idx] + msum[(size_t)r * D_DIM + d] + x[(size_t)r * D_DIM + d]) * (1.0f / 3.0f);
    eaH[idx] = (_Float16)v;
}

// --------------------------- WMMA GEMM (M x 128) @ Wt + b ------------------
// C[m][j] = sum_k A[m][k] * W[j][k] + bias[j]
// Template modes: UNIT: a = 0.5*(A[row] + A[rev[row]]) (packed f16);
//                 RELU: relu on output; OUT16: f16 output.
// Plain path stages the A strip with async global->LDS b128 (ASYNCcnt).
// Fragment layouts per CDNA5 ISA 7.12.2 (wave32):
//   A 16x32 f16: lane m=l&15, half h=l>>4; elems 0..7 -> K=32s+8h+e,
//                elems 8..15 -> K=32s+16+8h+(e-8)  (two contiguous 8xf16 runs)
//   B 32x16 f16: lane n=l&15, half h=l>>4; same K mapping as A (two 8xf16 runs)
//   C/D 16x16 f32: row = 8h + vgpr, col = l&15.

template <bool UNIT, bool RELU, bool OUT16>
__global__ __launch_bounds__(128)
void mes_gemm_wmma(const _Float16* __restrict__ A, const int* __restrict__ rev_idx,
                   const _Float16* __restrict__ W, const float* __restrict__ bias,
                   float* __restrict__ C, _Float16* __restrict__ C16,
                   int row_off, int Mloc) {
    __shared__ _Float16 wlds[128 * 128];          // 32 KB
    __shared__ _Float16 alds[4][16 * 128];        // 16 KB
    int t = threadIdx.x;

    // stage weight (128x128 f16 = 8192 dwords)
    {
        const unsigned* wsrc = (const unsigned*)W;
        unsigned* wdst = (unsigned*)wlds;
        #pragma unroll
        for (int i = 0; i < 64; ++i) wdst[t + 128 * i] = wsrc[t + 128 * i];
    }

    int wave = t >> 5, lane = t & 31;
    int nstrips = Mloc >> 4;
    int strip0 = blockIdx.x * 4 + wave;
    int strip = strip0 < nstrips ? strip0 : (nstrips - 1);   // clamp so staging is valid
    int rbase = strip << 4;

    _Float16* am = &alds[wave][0];
    if (!UNIT) {
        // pure copy of the 16x128 f16 strip: 4096 B = 8 iters x 32 lanes x 16 B,
        // issued as async global->LDS b128 transfers (ASYNCcnt).
        unsigned lbase = (unsigned)(unsigned long long)(size_t)am;
        unsigned long long gbase =
            (unsigned long long)(size_t)(A + (size_t)(row_off + rbase) * D_DIM);
        #pragma unroll
        for (int r = 0; r < 8; ++r) {
            unsigned ldst = lbase + (unsigned)(r * 512 + lane * 16);
            unsigned long long gsrc = gbase + (unsigned long long)(r * 512 + lane * 16);
            asm volatile("global_load_async_to_lds_b128 %0, %1, off"
                         :: "v"(ldst), "v"(gsrc) : "memory");
        }
        asm volatile("s_wait_asynccnt 0x0" ::: "memory");
    } else {
        const v4h half4 = {(_Float16)0.5f, (_Float16)0.5f, (_Float16)0.5f, (_Float16)0.5f};
        #pragma unroll
        for (int r = 0; r < 16; ++r) {
            int grow = row_off + rbase + r;
            v4h a = *(const v4h*)(A + (size_t)grow * D_DIM + lane * 4);
            v4h b = *(const v4h*)(A + (size_t)rev_idx[grow] * D_DIM + lane * 4);
            *(v4h*)(am + r * 128 + lane * 4) = (a + b) * half4;
        }
    }
    __syncthreads();

    int m = lane & 15, hh = lane >> 4;

    V16 afrag[4];
    #pragma unroll
    for (int s = 0; s < 4; ++s) {
        afrag[s].h[0] = *(const v8h*)(am + m * 128 + s * 32 + hh * 8);
        afrag[s].h[1] = *(const v8h*)(am + m * 128 + s * 32 + 16 + hh * 8);
    }

    bool store_ok = (strip0 < nstrips);          // wave-uniform
    #pragma unroll
    for (int nt = 0; nt < 8; ++nt) {
        v8f acc = {0.f, 0.f, 0.f, 0.f, 0.f, 0.f, 0.f, 0.f};
        #pragma unroll
        for (int s = 0; s < 4; ++s) {
            V16 b;
            const v8h* bp = (const v8h*)(wlds + (size_t)(nt * 16 + m) * 128 + s * 32 + hh * 8);
            b.h[0] = bp[0]; b.h[1] = bp[2];
            acc = __builtin_amdgcn_wmma_f32_16x16x32_f16(
                false, afrag[s].v, false, b.v, (short)0, acc, false, false);
        }
        if (store_ok) {
            int colj = nt * 16 + m;
            float bb = bias[colj];
            #pragma unroll
            for (int i = 0; i < 8; ++i) {
                float val = acc[i] + bb;
                if (RELU) val = fmaxf(val, 0.0f);
                size_t oidx = (size_t)(rbase + hh * 8 + i) * D_DIM + colj;
                if (OUT16) C16[oidx] = (_Float16)val;
                else       C[oidx] = val;
            }
        }
    }
}

// --------------------------- Q projection / score --------------------------

__global__ void mes_q_proj(const float* __restrict__ query, const float* __restrict__ qW,
                           const float* __restrict__ qb, float* __restrict__ Qn) {
    int i = blockIdx.x * blockDim.x + threadIdx.x;
    if (i >= B_GR * D_DIM) return;
    int b = i >> 7, j = i & 127;
    const float* qr = query + (size_t)b * D_DIM;
    const float* wr = qW + (size_t)j * D_DIM;
    float s = qb[j];
    for (int k = 0; k < D_DIM; ++k) s += qr[k] * wr[k];
    Qn[i] = s;
}

__global__ void mes_q_norm(const float* __restrict__ Qn, float* __restrict__ Qnorm) {
    int b = threadIdx.x;
    if (b >= B_GR) return;
    float s = 0.f;
    for (int j = 0; j < D_DIM; ++j) { float v = Qn[b * D_DIM + j]; s += v * v; }
    Qnorm[b] = sqrtf(s);
}

__global__ void mes_score(const float* __restrict__ Kch, const float* __restrict__ Qn,
                          const float* __restrict__ Qnorm, const int* __restrict__ row,
                          const int* __restrict__ batch, float* __restrict__ score,
                          int c0, int cn) {
    int i = blockIdx.x * blockDim.x + threadIdx.x;
    if (i >= cn) return;
    int e = c0 + i;
    int eb = batch[row[e]];
    const float* k = Kch + (size_t)i * D_DIM;
    const float* q = Qn + (size_t)eb * D_DIM;
    float dot = 0.f, nk = 0.f;
    for (int d = 0; d < D_DIM; ++d) { float kv = k[d]; dot += kv * q[d]; nk += kv * kv; }
    score[e] = tanhf(dot / (Qnorm[eb] * sqrtf(nk)));
}

__global__ void mes_graph_accum(const float* __restrict__ score, const int* __restrict__ row,
                                const int* __restrict__ batch, float* __restrict__ gs,
                                float* __restrict__ gc, int E) {
    int e = blockIdx.x * blockDim.x + threadIdx.x;
    if (e >= E) return;
    int b = batch[row[e]];
    atomicAdd(&gs[b], score[e]);
    atomicAdd(&gc[b], 1.0f);
}

__global__ void mes_perm(const float* __restrict__ score, const int* __restrict__ row,
                         const int* __restrict__ batch, const float* __restrict__ gs,
                         const float* __restrict__ gc, int* __restrict__ perm, int E) {
    int e = blockIdx.x * blockDim.x + threadIdx.x;
    if (e >= E) return;
    int b = batch[row[e]];
    float mean = gs[b] / fmaxf(gc[b], 1.0f);
    float s = score[e];
    perm[e] = (s < mean && s <= 0.0f) ? 1 : 0;
}

// --------------------------- ea2 = LN(PReLU(V*score)) (in place) -----------

__global__ void mes_ea2(float* __restrict__ V, const float* __restrict__ score,
                        const float* __restrict__ prelu_a, const float* __restrict__ ln_g,
                        const float* __restrict__ ln_b, int E) {
    int gw = (blockIdx.x * blockDim.x + threadIdx.x) >> 5;   // one wave per edge
    int lane = threadIdx.x & 31;
    if (gw >= E) return;
    float a = prelu_a[0];
    float s = score[gw];
    float* p = V + (size_t)gw * D_DIM;
    float v[4]; float sum = 0.f, sq = 0.f;
    for (int j = 0; j < 4; ++j) {
        float t = p[lane + 32 * j] * s;
        t = t >= 0.f ? t : a * t;
        v[j] = t; sum += t; sq += t * t;
    }
    for (int mofs = 16; mofs >= 1; mofs >>= 1) {
        sum += __shfl_xor(sum, mofs, 32);
        sq  += __shfl_xor(sq,  mofs, 32);
    }
    float mu = sum * (1.0f / 128.0f);
    float var = sq * (1.0f / 128.0f) - mu * mu;
    float inv = rsqrtf(var + LN_EPS);
    for (int j = 0; j < 4; ++j) {
        int d = lane + 32 * j;
        p[d] = (v[j] - mu) * inv * ln_g[d] + ln_b[d];
    }
}

// --------------------------- connected components --------------------------

__global__ void mes_cc_init(int* lab, int n) {
    int i = blockIdx.x * blockDim.x + threadIdx.x;
    if (i < n) lab[i] = i;
}
__global__ void mes_cc_edge(const int* __restrict__ labA, int* __restrict__ labB,
                            const int* __restrict__ row, const int* __restrict__ col,
                            const int* __restrict__ perm, int E) {
    int e = blockIdx.x * blockDim.x + threadIdx.x;
    if (e >= E || !perm[e]) return;
    atomicMin(&labB[row[e]], labA[col[e]]);
    atomicMin(&labB[col[e]], labA[row[e]]);
}
__global__ void mes_cc_jump(const int* __restrict__ in, int* __restrict__ out, int n) {
    int i = blockIdx.x * blockDim.x + threadIdx.x;
    if (i < n) out[i] = in[in[i]];
}

// --------------------------- GINE + pooling --------------------------------

__global__ void mes_gine_edge(const float* __restrict__ x, const float* __restrict__ ea2,
                              const int* __restrict__ row, const int* __restrict__ col,
                              const int* __restrict__ perm, float* __restrict__ agg, int E) {
    int idx = blockIdx.x * blockDim.x + threadIdx.x;
    int e = idx >> 7, d = idx & 127;
    if (e >= E || !perm[e]) return;
    float t = x[(size_t)row[e] * D_DIM + d] + ea2[(size_t)e * D_DIM + d];
    if (t > 0.0f) atomicAdd(&agg[(size_t)col[e] * D_DIM + d], t);
}
// h = x + agg, stored f16 (consumed only by WMMA GEMM)
__global__ void mes_add_f16(const float* __restrict__ a, const float* __restrict__ b,
                            _Float16* __restrict__ c, int n) {
    int i = blockIdx.x * blockDim.x + threadIdx.x;
    if (i < n) c[i] = (_Float16)(a[i] + b[i]);
}

__global__ void mes_logit(const float* __restrict__ xs, const float* __restrict__ attnW,
                          float* __restrict__ logit, int n) {
    int i = blockIdx.x * blockDim.x + threadIdx.x;
    if (i >= n) return;
    const float* p = xs + (size_t)i * D_DIM;
    float s = 0.f;
    for (int d = 0; d < D_DIM; ++d) s += p[d] * attnW[d];
    logit[i] = s > 0.f ? s : 0.01f * s;
}
__global__ void mes_seg_max(const float* __restrict__ logit, const int* __restrict__ comp,
                            float* __restrict__ mx, int n) {
    int i = blockIdx.x * blockDim.x + threadIdx.x;
    if (i < n) atomicMaxFloat(&mx[comp[i]], logit[i]);
}
__global__ void mes_exp_den(float* __restrict__ logit, const int* __restrict__ comp,
                            const float* __restrict__ mx, float* __restrict__ den, int n) {
    int i = blockIdx.x * blockDim.x + threadIdx.x;
    if (i >= n) return;
    float e = expf(logit[i] - mx[comp[i]]);
    logit[i] = e;
    atomicAdd(&den[comp[i]], e);
}
__global__ void mes_pool(const float* __restrict__ ex, const float* __restrict__ den,
                         const int* __restrict__ comp, const float* __restrict__ xs,
                         float* __restrict__ out_x, int n) {
    int idx = blockIdx.x * blockDim.x + threadIdx.x;
    int i = idx >> 7, d = idx & 127;
    if (i >= n) return;
    float a = ex[i] / den[comp[i]];
    atomicAdd(&out_x[(size_t)comp[i] * D_DIM + d], a * xs[(size_t)i * D_DIM + d]);
}

// --------------------------- coalesce (radix sort + scan) ------------------

__global__ void mes_make_keys(const int* __restrict__ comp, const int* __restrict__ row,
                              const int* __restrict__ col, unsigned* __restrict__ keys,
                              unsigned* __restrict__ vals, int E) {
    int e = blockIdx.x * blockDim.x + threadIdx.x;
    if (e >= E) return;
    int rc = comp[row[e]], cc = comp[col[e]];
    keys[e] = (rc != cc) ? ((unsigned)rc * (unsigned)N_NODE + (unsigned)cc) : INV_KEY;
    vals[e] = (unsigned)e;
}

#define RCHUNK 4096
__global__ void mes_radix_hist(const unsigned* __restrict__ keys, int n, int shift,
                               unsigned* __restrict__ gHist, int numBlocks) {
    __shared__ unsigned h[256];
    h[threadIdx.x] = 0;
    __syncthreads();
    int base = blockIdx.x * RCHUNK;
    for (int i = threadIdx.x; i < RCHUNK; i += 256) {
        int idx = base + i;
        if (idx < n) atomicAdd(&h[(keys[idx] >> shift) & 255u], 1u);
    }
    __syncthreads();
    gHist[threadIdx.x * numBlocks + blockIdx.x] = h[threadIdx.x];
}
__global__ void mes_radix_scan(unsigned* gHist, int total) {
    if (threadIdx.x == 0 && blockIdx.x == 0) {
        unsigned run = 0;
        for (int i = 0; i < total; ++i) { unsigned v = gHist[i]; gHist[i] = run; run += v; }
    }
}
__global__ void mes_radix_scatter(const unsigned* __restrict__ keys, const unsigned* __restrict__ vals,
                                  unsigned* __restrict__ keysOut, unsigned* __restrict__ valsOut,
                                  int n, int shift, const unsigned* __restrict__ gHist, int numBlocks) {
    __shared__ unsigned off[256];
    off[threadIdx.x] = gHist[threadIdx.x * numBlocks + blockIdx.x];
    __syncthreads();
    if (threadIdx.x == 0) {
        int base = blockIdx.x * RCHUNK;
        int end = base + RCHUNK; if (end > n) end = n;
        for (int i = base; i < end; ++i) {
            unsigned k = keys[i];
            unsigned b = (k >> shift) & 255u;
            unsigned p = off[b]++;
            keysOut[p] = k;
            valsOut[p] = vals[i];
        }
    }
}

__global__ void mes_flags(const unsigned* __restrict__ keys, int* __restrict__ flags, int n) {
    int i = blockIdx.x * blockDim.x + threadIdx.x;
    if (i < n) flags[i] = (i > 0 && keys[i] != keys[i - 1]) ? 1 : 0;
}
__global__ void mes_scan_block(int* __restrict__ data, int* __restrict__ bsums, int n) {
    __shared__ int s[512];
    int i = blockIdx.x * 512 + threadIdx.x;
    int v = (i < n) ? data[i] : 0;
    s[threadIdx.x] = v;
    __syncthreads();
    for (int off = 1; off < 512; off <<= 1) {
        int t = (threadIdx.x >= (unsigned)off) ? s[threadIdx.x - off] : 0;
        __syncthreads();
        s[threadIdx.x] += t;
        __syncthreads();
    }
    if (i < n) data[i] = s[threadIdx.x];
    if (threadIdx.x == 511) bsums[blockIdx.x] = s[511];
}
__global__ void mes_scan_sums(int* bsums, int nb) {
    if (threadIdx.x == 0 && blockIdx.x == 0) {
        int run = 0;
        for (int i = 0; i < nb; ++i) { int v = bsums[i]; bsums[i] = run; run += v; }
    }
}
__global__ void mes_scan_add(int* __restrict__ data, const int* __restrict__ bsums, int n) {
    int i = blockIdx.x * 512 + threadIdx.x;
    if (i < n) data[i] += bsums[blockIdx.x];
}

__global__ void mes_seg_outputs(const unsigned* __restrict__ keys, const int* __restrict__ seg,
                                int* __restrict__ rowi, int* __restrict__ coli, int n) {
    int i = blockIdx.x * blockDim.x + threadIdx.x;
    if (i >= n) return;
    unsigned k = keys[i];
    int s = seg[i];
    if (k != INV_KEY) { rowi[s] = (int)(k / (unsigned)N_NODE); coli[s] = (int)(k % (unsigned)N_NODE); }
    else              { rowi[s] = -1; coli[s] = -1; }
}
__global__ void mes_attr_out(const unsigned* __restrict__ keys, const unsigned* __restrict__ vals,
                             const int* __restrict__ seg, const float* __restrict__ ea2,
                             float* __restrict__ out_ea, int n) {
    int idx = blockIdx.x * blockDim.x + threadIdx.x;
    int i = idx >> 7, d = idx & 127;
    if (i >= n || keys[i] == INV_KEY) return;
    atomicAdd(&out_ea[(size_t)seg[i] * D_DIM + d], ea2[(size_t)vals[i] * D_DIM + d]);
}
__global__ void mes_ei_convert(const int* __restrict__ rowi, const int* __restrict__ coli,
                               float* __restrict__ out_ei, int E) {
    int s = blockIdx.x * blockDim.x + threadIdx.x;
    if (s >= E) return;
    out_ei[s] = (float)rowi[s];
    out_ei[E + s] = (float)coli[s];
}

// ---------------------------------------------------------------------------

extern "C" void kernel_launch(void* const* d_in, const int* in_sizes, int n_in,
                              void* d_out, int out_size, void* d_ws, size_t ws_size,
                              hipStream_t stream) {
    const int D = D_DIM, N = N_NODE, E = E_EDGE, B = B_GR;

    const float* x       = (const float*)d_in[0];
    const float* eattr   = (const float*)d_in[1];
    const float* query   = (const float*)d_in[2];
    const float* qW      = (const float*)d_in[3];
    const float* qb      = (const float*)d_in[4];
    const float* kW      = (const float*)d_in[5];
    const float* kb      = (const float*)d_in[6];
    const float* vW      = (const float*)d_in[7];
    const float* vb      = (const float*)d_in[8];
    const float* prelu_a = (const float*)d_in[9];
    const float* ln_g    = (const float*)d_in[10];
    const float* ln_b    = (const float*)d_in[11];
    const float* attnW   = (const float*)d_in[12];
    const float* g1W     = (const float*)d_in[13];
    const float* g1b     = (const float*)d_in[14];
    const float* g2W     = (const float*)d_in[15];
    const float* g2b     = (const float*)d_in[16];
    const int* edge_index= (const int*)d_in[17];
    const int* batch     = (const int*)d_in[18];
    const int* rev_idx   = (const int*)d_in[19];
    const int* row = edge_index;
    const int* col = edge_index + E;

    float* out_x  = (float*)d_out;                 // N*D
    float* out_ei = out_x + (size_t)N * D;         // 2*E
    float* out_ea = out_ei + 2 * (size_t)E;        // E*D
    float* out_sc = out_ea + (size_t)E * D;        // E

    // ---------------- workspace layout ----------------
    char* base = (char*)d_ws;
    size_t off = 0;
    auto alloc = [&](size_t bytes) { char* p = base + off; off = (off + bytes + 255) & ~(size_t)255; return p; };

    _Float16* eaH = (_Float16*)alloc((size_t)E * D * 2);  // f16 ea; later reused for node phase
    float* Vbuf  = (float*)alloc((size_t)E * D * 4);      // V, then ea2 in place
    float* Kch   = (float*)alloc((size_t)K_CHUNK * D * 4);
    float* msum  = (float*)alloc((size_t)N * D * 4);
    float* cnt   = (float*)alloc((size_t)N * 4);
    float* Qn    = (float*)alloc((size_t)B * D * 4);
    float* Qnorm = (float*)alloc((size_t)B * 4);
    float* gs    = (float*)alloc((size_t)B * 4);
    float* gc    = (float*)alloc((size_t)B * 4);
    int*   perm  = (int*)alloc((size_t)E * 4);
    int*   labA  = (int*)alloc((size_t)N * 4);
    int*   labB  = (int*)alloc((size_t)N * 4);
    int*   labC  = (int*)alloc((size_t)N * 4);
    float* logit = (float*)alloc((size_t)N * 4);
    float* mx    = (float*)alloc((size_t)N * 4);
    float* den   = (float*)alloc((size_t)N * 4);
    _Float16* wfV  = (_Float16*)alloc(128 * 128 * 2);
    _Float16* wfK  = (_Float16*)alloc(128 * 128 * 2);
    _Float16* wfG1 = (_Float16*)alloc(128 * 128 * 2);
    _Float16* wfG2 = (_Float16*)alloc(128 * 128 * 2);
    const int RNB = cdiv(E, RCHUNK);                       // 98
    unsigned* hist = (unsigned*)alloc((size_t)RNB * 256 * 4);
    const int SNB = cdiv(E, 512);                          // 782
    int* bsums = (int*)alloc((size_t)SNB * 4);

    // buffers carved from the (dead after K-phase) eaH region (102.4 MB)
    char* eabase = (char*)eaH;
    float*     agg   = (float*)(eabase);                                   // N*D f32
    _Float16*  h16   = (_Float16*)(eabase + (size_t)N * D * 4);            // N*D f16
    _Float16*  hid16 = (_Float16*)(eabase + (size_t)N * D * 6);            // N*D f16
    float*     xsp   = (float*)(eabase + (size_t)N * D * 8);               // N*D f32
    char* sbase = eabase + (size_t)N * D * 12;
    unsigned* keysA = (unsigned*)(sbase);
    unsigned* keysB = (unsigned*)(sbase + (size_t)E * 4);
    unsigned* valsA = (unsigned*)(sbase + 2 * (size_t)E * 4);
    unsigned* valsB = (unsigned*)(sbase + 3 * (size_t)E * 4);
    int* seg  = (int*)(sbase + 4 * (size_t)E * 4);
    int* rowi = (int*)(sbase + 5 * (size_t)E * 4);
    int* coli = (int*)(sbase + 6 * (size_t)E * 4);

    const int T = 256;
    // ---------------- stage 1: msg = seg_mean(edge_attr, col); ea ----------
    mes_fill_f<<<cdiv(N * D, T), T, 0, stream>>>(msum, 0.f, N * D);
    mes_fill_f<<<cdiv(N, T), T, 0, stream>>>(cnt, 0.f, N);
    mes_edge_seg_sum<<<cdiv(E * D, T), T, 0, stream>>>(eattr, col, msum, cnt, E);
    mes_msg_div<<<cdiv(N * D, T), T, 0, stream>>>(msum, cnt, N);
    mes_ea<<<cdiv(E * D, T), T, 0, stream>>>(eattr, msum, x, row, eaH, E);

    // ---------------- f16 weights ------------------------------------------
    mes_cvt_f16<<<cdiv(128 * 128, T), T, 0, stream>>>(vW, wfV, 128 * 128);
    mes_cvt_f16<<<cdiv(128 * 128, T), T, 0, stream>>>(kW, wfK, 128 * 128);
    mes_cvt_f16<<<cdiv(128 * 128, T), T, 0, stream>>>(g1W, wfG1, 128 * 128);
    mes_cvt_f16<<<cdiv(128 * 128, T), T, 0, stream>>>(g2W, wfG2, 128 * 128);

    // ---------------- V = ea @ vW^T + vb (WMMA, async A-staging) -----------
    mes_gemm_wmma<false, false, false><<<cdiv(E / 16, 4), 128, 0, stream>>>(
        eaH, nullptr, wfV, vb, Vbuf, nullptr, 0, E);

    // ---------------- Q projection -----------------------------------------
    mes_q_proj<<<cdiv(B * D, T), T, 0, stream>>>(query, qW, qb, Qn);
    mes_q_norm<<<1, 64, 0, stream>>>(Qn, Qnorm);

    // ---------------- K = unit @ kW^T + kb (WMMA, chunked) + score ---------
    for (int c = 0; c < E / K_CHUNK; ++c) {
        int c0 = c * K_CHUNK;
        mes_gemm_wmma<true, false, false><<<cdiv(K_CHUNK / 16, 4), 128, 0, stream>>>(
            eaH, rev_idx, wfK, kb, Kch, nullptr, c0, K_CHUNK);
        mes_score<<<cdiv(K_CHUNK, T), T, 0, stream>>>(Kch, Qn, Qnorm, row, batch, out_sc, c0, K_CHUNK);
    }

    // ---------------- mean score per graph + perm mask ---------------------
    mes_fill_f<<<1, T, 0, stream>>>(gs, 0.f, B);
    mes_fill_f<<<1, T, 0, stream>>>(gc, 0.f, B);
    mes_graph_accum<<<cdiv(E, T), T, 0, stream>>>(out_sc, row, batch, gs, gc, E);
    mes_perm<<<cdiv(E, T), T, 0, stream>>>(out_sc, row, batch, gs, gc, perm, E);

    // ---------------- ea2 = LN(PReLU(V * score)) in place ------------------
    mes_ea2<<<cdiv(E * 32, T), T, 0, stream>>>(Vbuf, out_sc, prelu_a, ln_g, ln_b, E);
    float* ea2 = Vbuf;

    // ---------------- connected components (min-label + pointer jumping) ---
    mes_cc_init<<<cdiv(N, T), T, 0, stream>>>(labA, N);
    for (int it = 0; it < 20; ++it) {
        hipMemcpyAsync(labB, labA, (size_t)N * 4, hipMemcpyDeviceToDevice, stream);
        mes_cc_edge<<<cdiv(E, T), T, 0, stream>>>(labA, labB, row, col, perm, E);
        mes_cc_jump<<<cdiv(N, T), T, 0, stream>>>(labB, labC, N);
        mes_cc_jump<<<cdiv(N, T), T, 0, stream>>>(labC, labA, N);
    }
    int* comp = labA;

    // ---------------- GINE conv + node MLP (WMMA) --------------------------
    mes_fill_f<<<cdiv(N * D, T), T, 0, stream>>>(agg, 0.f, N * D);
    mes_gine_edge<<<cdiv(E * D, T), T, 0, stream>>>(x, ea2, row, col, perm, agg, E);
    mes_add_f16<<<cdiv(N * D, T), T, 0, stream>>>(x, agg, h16, N * D);
    mes_gemm_wmma<false, true, true><<<cdiv(N / 16, 4), 128, 0, stream>>>(
        h16, nullptr, wfG1, g1b, nullptr, hid16, 0, N);
    mes_gemm_wmma<false, false, false><<<cdiv(N / 16, 4), 128, 0, stream>>>(
        hid16, nullptr, wfG2, g2b, xsp, nullptr, 0, N);

    // ---------------- per-component attention softmax pooling --------------
    mes_logit<<<cdiv(N, T), T, 0, stream>>>(xsp, attnW, logit, N);
    mes_fill_f<<<cdiv(N, T), T, 0, stream>>>(mx, -3.402823466e38f, N);
    mes_fill_f<<<cdiv(N, T), T, 0, stream>>>(den, 0.f, N);
    mes_fill_f<<<cdiv(N * D, T), T, 0, stream>>>(out_x, 0.f, N * D);
    mes_seg_max<<<cdiv(N, T), T, 0, stream>>>(logit, comp, mx, N);
    mes_exp_den<<<cdiv(N, T), T, 0, stream>>>(logit, comp, mx, den, N);
    mes_pool<<<cdiv(N * D, T), T, 0, stream>>>(logit, den, comp, xsp, out_x, N);

    // ---------------- edge reduce + coalesce -------------------------------
    mes_make_keys<<<cdiv(E, T), T, 0, stream>>>(comp, row, col, keysA, valsA, E);
    for (int pass = 0; pass < 4; ++pass) {
        unsigned* sk = (pass & 1) ? keysB : keysA;
        unsigned* sv = (pass & 1) ? valsB : valsA;
        unsigned* dk = (pass & 1) ? keysA : keysB;
        unsigned* dv = (pass & 1) ? valsA : valsB;
        mes_radix_hist<<<RNB, 256, 0, stream>>>(sk, E, pass * 8, hist, RNB);
        mes_radix_scan<<<1, 1, 0, stream>>>(hist, RNB * 256);
        mes_radix_scatter<<<RNB, 256, 0, stream>>>(sk, sv, dk, dv, E, pass * 8, hist, RNB);
    }
    // 4 passes -> sorted result back in keysA/valsA
    mes_flags<<<cdiv(E, T), T, 0, stream>>>(keysA, seg, E);
    mes_scan_block<<<SNB, 512, 0, stream>>>(seg, bsums, E);
    mes_scan_sums<<<1, 1, 0, stream>>>(bsums, SNB);
    mes_scan_add<<<SNB, 512, 0, stream>>>(seg, bsums, E);

    mes_fill_i<<<cdiv(E, T), T, 0, stream>>>(rowi, INT_MIN, E);
    mes_fill_i<<<cdiv(E, T), T, 0, stream>>>(coli, INT_MIN, E);
    mes_fill_f<<<cdiv(E * D, T), T, 0, stream>>>(out_ea, 0.f, E * D);
    mes_seg_outputs<<<cdiv(E, T), T, 0, stream>>>(keysA, seg, rowi, coli, E);
    mes_attr_out<<<cdiv(E * D, T), T, 0, stream>>>(keysA, valsA, seg, ea2, out_ea, E);
    mes_ei_convert<<<cdiv(E, T), T, 0, stream>>>(rowi, coli, out_ei, E);

    (void)in_sizes; (void)n_in; (void)out_size; (void)ws_size;
}
